// LocalSelfAttention_78606491452235
// MI455X (gfx1250) — matrix operand order repaired and checked by reading, compiled, not run on hardware
//
#include <hip/hip_runtime.h>
#include <hip/hip_bf16.h>
#include <stdint.h>

// LocalSelfAttention for MI455X (gfx1250, wave32, WMMA).
// Pipeline: [qkv GEMM f32->f16 + scatter] -> [flash windowed attention] -> [out GEMM].
// All matmuls use v_wmma_f32_16x16x32_f16; staging uses global_load_async_to_lds_b128.
// All WMMA B-operands are kept n-major in LDS so fragments load as ds_load_b128.
// Workspace (halves): q | k | vT | ctx, each 4*8192*512 = 16,777,216 (total 128 MB).
//   q,k : [b][nb][h][w][dh]   (row-major per head)
//   vT  : [b][nb][h][dh][w]   (transposed per head, so P@V B-frags are contiguous)

typedef __attribute__((ext_vector_type(16))) _Float16 v16h;
typedef __attribute__((ext_vector_type(8)))  _Float16 v8h;
typedef __attribute__((ext_vector_type(8)))  float    v8f;
typedef __attribute__((ext_vector_type(4)))  int      v4i;

#define DEV __device__ __attribute__((always_inline)) static inline

#if defined(__HIP_DEVICE_COMPILE__) && __has_builtin(__builtin_amdgcn_global_load_async_to_lds_b128)
#define HAVE_ASYNC_LDS 1
#else
#define HAVE_ASYNC_LDS 0
#endif

DEV void async_copy16(void* lds_dst, const void* gsrc) {
#if HAVE_ASYNC_LDS
  __builtin_amdgcn_global_load_async_to_lds_b128(
      (__attribute__((address_space(1))) v4i*)gsrc,
      (__attribute__((address_space(3))) v4i*)lds_dst, 0, 0);
#else
  *(uint4*)lds_dst = *(const uint4*)gsrc;
#endif
}

DEV void async_join() {
#if HAVE_ASYNC_LDS
#if __has_builtin(__builtin_amdgcn_s_wait_asynccnt)
  __builtin_amdgcn_s_wait_asynccnt(0);
#else
  asm volatile("s_wait_asynccnt 0" ::: "memory");
#endif
#endif
}

// A fragment (16x32 f16, row-major src, leading dim ld in halves).
// Lane L<16: M=L, K in {0..7}|{16..23}; lane L>=16: M=L-16, K in {8..15}|{24..31}.
// Per lane: two contiguous 16-byte runs -> 2x ds_load_b128.
DEV v16h load_a_frag(const _Float16* src, int ld) {
  const int lane = threadIdx.x & 31;
  const int m  = lane & 15;
  const int kb = (lane >> 4) << 3;
  v16h a;
#pragma unroll
  for (int p = 0; p < 8; ++p) {
    const int k = ((p & 4) << 2) + kb + ((p & 3) << 1);
    a[2*p]   = src[m*ld + k];
    a[2*p+1] = src[m*ld + k + 1];
  }
  return a;
}

// B fragment (32x16 f16) from an n-major source: element (k,n) at src[n*ld + k].
// Lane n=lane&15 reads K in {kb..kb+7}|{16+kb..16+kb+7}, kb = (lane>=16 ? 8 : 0):
// two 16-byte vector loads -> 2x ds_load_b128.
DEV v16h load_b_frag_c(const _Float16* src, int ld) {
  const int lane = threadIdx.x & 31;
  const int n  = lane & 15;
  const int kb = (lane >> 4) << 3;
  const v8h lo = *(const v8h*)(src + n * ld + kb);
  const v8h hi = *(const v8h*)(src + n * ld + 16 + kb);
  return __builtin_shufflevector(lo, hi, 0,1,2,3,4,5,6,7,8,9,10,11,12,13,14,15);
}

DEV v8f wmma_f16(v16h a, v16h b, v8f c) {
  return __builtin_amdgcn_wmma_f32_16x16x32_f16(false, a, false, b, (short)0, c, false, false);
}

DEV v8f v8f_zero() {
  v8f z;
#pragma unroll
  for (int e = 0; e < 8; ++e) z[e] = 0.f;
  return z;
}

// ---------------------------------------------------------------------------
// Kernel 1: qkv = x @ Wqkv + bqkv ; scatter q,k -> [b][nb][h][w][dh],
// v -> transposed [b][nb][h][dh][w] (packed 16B stores).
// Block tile 128(M) x 64(N), 8 waves (4x2), each wave 32x32 (2x2 WMMA tiles).
// ---------------------------------------------------------------------------
__global__ __launch_bounds__(256) void qkv_kernel(
    const float* __restrict__ x, const float* __restrict__ Wqkv,
    const float* __restrict__ bqkv,
    _Float16* __restrict__ qO, _Float16* __restrict__ kO, _Float16* __restrict__ vO)
{
  __shared__ __align__(32) _Float16 As[128 * 32];   // row-major (m,k)
  __shared__ __align__(32) _Float16 Bt[64 * 32];    // n-major   (n,k)

  const int tid  = threadIdx.x;
  const int lane = tid & 31;
  const int wave = tid >> 5;
  const int wy = wave >> 1, wx = wave & 1;
  const int row0 = blockIdx.x * 128;
  const int col0 = blockIdx.y * 64;

  v8f acc[2][2];
#pragma unroll
  for (int i = 0; i < 2; ++i)
#pragma unroll
    for (int j = 0; j < 2; ++j) acc[i][j] = v8f_zero();

  for (int kk = 0; kk < 512; kk += 32) {
    __syncthreads();
    {
      // Stage A: x tile 128x32 f32 -> f16 (2 threads per row, 16 halves each)
      const int r  = tid >> 1;
      const int c0 = (tid & 1) * 16;
      const float* sA = x + (size_t)(row0 + r) * 512 + kk + c0;
#pragma unroll
      for (int q4 = 0; q4 < 4; ++q4) {
        float4 f = *(const float4*)(sA + 4 * q4);
        _Float16* d = &As[r * 32 + c0 + 4 * q4];
        d[0] = (_Float16)f.x; d[1] = (_Float16)f.y;
        d[2] = (_Float16)f.z; d[3] = (_Float16)f.w;
      }
      // Stage B transposed: Wqkv tile 32(k)x64(n) f32 -> f16 n-major
      const int r2 = tid >> 3;            // k row 0..31
      const int c2 = (tid & 7) * 8;       // n col base
      const float* sB = Wqkv + (size_t)(kk + r2) * 1536 + col0 + c2;
      float4 f0 = *(const float4*)(sB);
      float4 f1 = *(const float4*)(sB + 4);
      _Float16 hv[8] = {(_Float16)f0.x, (_Float16)f0.y, (_Float16)f0.z, (_Float16)f0.w,
                        (_Float16)f1.x, (_Float16)f1.y, (_Float16)f1.z, (_Float16)f1.w};
#pragma unroll
      for (int q8 = 0; q8 < 8; ++q8)
        Bt[(c2 + q8) * 32 + r2] = hv[q8];
    }
    __syncthreads();

    v16h a0 = load_a_frag(&As[(wy * 32) * 32], 32);
    v16h a1 = load_a_frag(&As[(wy * 32 + 16) * 32], 32);
    v16h b0 = load_b_frag_c(&Bt[(wx * 32) * 32], 32);
    v16h b1 = load_b_frag_c(&Bt[(wx * 32 + 16) * 32], 32);
    acc[0][0] = wmma_f16(a0, b0, acc[0][0]);
    acc[0][1] = wmma_f16(a0, b1, acc[0][1]);
    acc[1][0] = wmma_f16(a1, b0, acc[1][0]);
    acc[1][1] = wmma_f16(a1, b1, acc[1][1]);
  }

  // Epilogue: bias add, split into q/k/v, scatter.
  const int halfadd = (lane >> 4) << 3;
  const int ncol    = lane & 15;
#pragma unroll
  for (int ft = 0; ft < 2; ++ft)
#pragma unroll
    for (int fn = 0; fn < 2; ++fn) {
      const int cg   = col0 + wx * 32 + fn * 16 + ncol;  // 0..1535
      const int part = cg >> 9;                          // 0=q 1=k 2=v
      const int hh   = (cg >> 6) & 7;
      const int dd   = cg & 63;
      const float bias = bqkv[cg];
      const int m0 = row0 + wy * 32 + ft * 16 + halfadd; // row of e=0 (multiple of 8)
      const int bb = m0 >> 13;
      const int nn = (m0 >> 8) & 31;
      const int ww = m0 & 255;
      const size_t head = ((size_t)(bb * 32 + nn) * 8 + hh);
      if (part == 2) {
        // v transposed: [head][dh][w]; 8 consecutive w per lane -> one 16B store
        v8h pk;
#pragma unroll
        for (int e = 0; e < 8; ++e) pk[e] = (_Float16)(acc[ft][fn][e] + bias);
        *(v8h*)(vO + (head * 64 + dd) * 256 + ww) = pk;
      } else {
        _Float16* dst = (part == 0) ? qO : kO;
#pragma unroll
        for (int e = 0; e < 8; ++e)
          dst[(head * 256 + ww + e) * 64 + dd] = (_Float16)(acc[ft][fn][e] + bias);
      }
    }
}

// ---------------------------------------------------------------------------
// Kernel 2: windowed attention, flash-style online softmax.
// One block per (b, nb, h). 8 waves, 32 query rows each. Key chunks of 64
// over 2W=512 candidate slots (prev block + current block).
// ---------------------------------------------------------------------------
__global__ __launch_bounds__(256) void attn_kernel(
    const _Float16* __restrict__ qB, const _Float16* __restrict__ kB,
    const _Float16* __restrict__ vT, const float* __restrict__ relb,
    _Float16* __restrict__ ctx)
{
  __shared__ __align__(32) _Float16 Kc[64 * 64];     // key chunk   (j, d) = n-major for QK^T
  __shared__ __align__(32) _Float16 Vt[64 * 64];     // value chunk (d, j) = n-major for P@V
  __shared__ __align__(32) float    biasS[256];      // rel_pos_bias[:, h]
  __shared__ __align__(32) _Float16 Qs[256 * 64];    // Q tile; reused as per-wave P scratch

  const int tid  = threadIdx.x;
  const int lane = tid & 31;
  const int wave = tid >> 5;
  const int blk = blockIdx.x;
  const int h = blk & 7;
  const int n = (blk >> 3) & 31;
  const int b = blk >> 8;

  biasS[tid] = relb[tid * 8 + h];

  const size_t headBase = ((size_t)((b * 32 + n) * 8 + h)) * (256 * 64);
  for (int i = tid; i < 2048; i += 256)
    async_copy16(&Qs[i * 8], &qB[headBase + (size_t)i * 8]);
  async_join();
  __syncthreads();

  const int u0 = wave * 32;
  v16h qf[2][2];
#pragma unroll
  for (int t = 0; t < 2; ++t)
#pragma unroll
    for (int ks = 0; ks < 2; ++ks)
      qf[t][ks] = load_a_frag(&Qs[(u0 + t * 16) * 64 + ks * 32], 64);
  __syncthreads();                       // Qs now reusable as P scratch
  _Float16* Pw = &Qs[wave * (32 * 64)];  // 32x64 f16 per wave

  v8f  O[2][4];
  float m_s[2][8], l_s[2][8];
#pragma unroll
  for (int t = 0; t < 2; ++t) {
#pragma unroll
    for (int dt = 0; dt < 4; ++dt) O[t][dt] = v8f_zero();
#pragma unroll
    for (int e = 0; e < 8; ++e) { m_s[t][e] = -1e30f; l_s[t][e] = 0.f; }
  }

  const float scale   = 0.125f;          // 1/sqrt(64)
  const int   halfadd = (lane >> 4) << 3;
  const int   ncol    = lane & 15;

  for (int c = 0; c < 8; ++c) {
    const int j0 = c * 64;
    const int bs = (j0 >= 256) ? n : (n - 1);   // source block (prev / current)
    __syncthreads();                             // previous chunk fully consumed
    if (bs >= 0) {
      const size_t head = ((size_t)((b * 32 + bs) * 8 + h)) * (256 * 64);
      const int jr = j0 & 255;
      for (int i = tid; i < 512; i += 256) {
        // K: rows j (64) x 64 d-halves, contiguous per row
        const int rj = i >> 3, c8 = (i & 7) * 8;
        async_copy16(&Kc[rj * 64 + c8], &kB[head + (size_t)(jr + rj) * 64 + c8]);
        // V^T: rows d (64) x 64 j-halves, row stride 256 in workspace
        async_copy16(&Vt[rj * 64 + c8], &vT[head + (size_t)rj * 256 + jr + c8]);
      }
      async_join();
    }
    __syncthreads();

    // Band for this wave: valid j in [u0+1, u0+31+256]; block 0 has no prev keys.
    if (!((bs >= 0) && (j0 + 63 >= u0 + 1) && (j0 <= u0 + 287))) continue;

#pragma unroll
    for (int t = 0; t < 2; ++t) {
      // S = Q @ K^T for this 16-row tile over the 64-key chunk
      v8f S[4];
#pragma unroll
      for (int jc = 0; jc < 4; ++jc) {
        v8f sc = v8f_zero();
#pragma unroll
        for (int ks = 0; ks < 2; ++ks) {
          v16h bf = load_b_frag_c(&Kc[jc * 16 * 64 + ks * 32], 64);  // B[k=d][n=j]
          sc = wmma_f16(qf[t][ks], bf, sc);
        }
        S[jc] = sc;
      }
      // scale + rel-pos bias + band/seq-start mask; chunk row-max
      float mc[8];
#pragma unroll
      for (int e = 0; e < 8; ++e) mc[e] = -1e30f;
#pragma unroll
      for (int jc = 0; jc < 4; ++jc)
#pragma unroll
        for (int e = 0; e < 8; ++e) {
          const int u = u0 + t * 16 + e + halfadd;
          const int j = j0 + jc * 16 + ncol;
          const int w = j - u - 1;
          const bool vld = (w >= 0) && (w < 256) && ((n > 0) || (j >= 256));
          const float s = vld ? (S[jc][e] * scale + biasS[w & 255]) : -1e30f;
          S[jc][e] = s;
          mc[e] = fmaxf(mc[e], s);
        }
#pragma unroll
      for (int off = 1; off < 16; off <<= 1)
#pragma unroll
        for (int e = 0; e < 8; ++e)
          mc[e] = fmaxf(mc[e], __shfl_xor(mc[e], off, 32));
      // online softmax update
      float alpha[8], rs[8];
#pragma unroll
      for (int e = 0; e < 8; ++e) {
        const float mn = fmaxf(m_s[t][e], mc[e]);
        alpha[e] = __expf(m_s[t][e] - mn);
        m_s[t][e] = mn;
        rs[e] = 0.f;
      }
#pragma unroll
      for (int jc = 0; jc < 4; ++jc)
#pragma unroll
        for (int e = 0; e < 8; ++e) {
          const float p = __expf(S[jc][e] - m_s[t][e]);
          S[jc][e] = p;
          rs[e] += p;
        }
#pragma unroll
      for (int off = 1; off < 16; off <<= 1)
#pragma unroll
        for (int e = 0; e < 8; ++e)
          rs[e] += __shfl_xor(rs[e], off, 32);
#pragma unroll
      for (int e = 0; e < 8; ++e)
        l_s[t][e] = l_s[t][e] * alpha[e] + rs[e];
#pragma unroll
      for (int dt = 0; dt < 4; ++dt)
#pragma unroll
        for (int e = 0; e < 8; ++e)
          O[t][dt][e] *= alpha[e];
      // P -> LDS (f16), re-fragment as WMMA A, accumulate O += P @ V
#pragma unroll
      for (int jc = 0; jc < 4; ++jc)
#pragma unroll
        for (int e = 0; e < 8; ++e)
          Pw[(t * 16 + e + halfadd) * 64 + jc * 16 + ncol] = (_Float16)S[jc][e];
#pragma unroll
      for (int ks = 0; ks < 2; ++ks) {
        v16h af = load_a_frag(&Pw[t * 16 * 64 + ks * 32], 64);
#pragma unroll
        for (int dt = 0; dt < 4; ++dt) {
          v16h bf = load_b_frag_c(&Vt[dt * 16 * 64 + ks * 32], 64);  // B[k=j][n=d]
          O[t][dt] = wmma_f16(af, bf, O[t][dt]);
        }
      }
    }
  }

  // Epilogue: normalize, write ctx f16 row-major [32768][512] (col = h*64+d)
#pragma unroll
  for (int t = 0; t < 2; ++t) {
    float inv[8];
#pragma unroll
    for (int e = 0; e < 8; ++e) inv[e] = 1.f / l_s[t][e];
#pragma unroll
    for (int dt = 0; dt < 4; ++dt)
#pragma unroll
      for (int e = 0; e < 8; ++e) {
        const int mlocal = u0 + t * 16 + e + halfadd;
        const size_t row = (size_t)b * 8192 + (size_t)n * 256 + mlocal;
        const int col = h * 64 + dt * 16 + ncol;
        ctx[row * 512 + col] = (_Float16)(O[t][dt][e] * inv[e]);
      }
  }
}

// ---------------------------------------------------------------------------
// Kernel 3: out = ctx @ Wo + bo (fp32 out). Same skeleton as kernel 1,
// ctx staged via async global->LDS b128 (already f16), Wo staged transposed.
// ---------------------------------------------------------------------------
__global__ __launch_bounds__(256) void proj_kernel(
    const _Float16* __restrict__ ctx, const float* __restrict__ Wo,
    const float* __restrict__ bo, float* __restrict__ out)
{
  __shared__ __align__(32) _Float16 As[128 * 32];   // row-major (m,k)
  __shared__ __align__(32) _Float16 Bt[64 * 32];    // n-major   (n,k)

  const int tid  = threadIdx.x;
  const int lane = tid & 31;
  const int wave = tid >> 5;
  const int wy = wave >> 1, wx = wave & 1;
  const int row0 = blockIdx.x * 128;
  const int col0 = blockIdx.y * 64;

  v8f acc[2][2];
#pragma unroll
  for (int i = 0; i < 2; ++i)
#pragma unroll
    for (int j = 0; j < 2; ++j) acc[i][j] = v8f_zero();

  for (int kk = 0; kk < 512; kk += 32) {
    __syncthreads();
    {
      for (int i = tid; i < 512; i += 256) {
        const int r  = i >> 2;
        const int c8 = (i & 3) * 8;
        async_copy16(&As[r * 32 + c8], &ctx[(size_t)(row0 + r) * 512 + kk + c8]);
      }
      const int r2 = tid >> 3;            // k row 0..31
      const int c2 = (tid & 7) * 8;       // n col base
      const float* sB = Wo + (size_t)(kk + r2) * 512 + col0 + c2;
      float4 f0 = *(const float4*)(sB);
      float4 f1 = *(const float4*)(sB + 4);
      _Float16 hv[8] = {(_Float16)f0.x, (_Float16)f0.y, (_Float16)f0.z, (_Float16)f0.w,
                        (_Float16)f1.x, (_Float16)f1.y, (_Float16)f1.z, (_Float16)f1.w};
#pragma unroll
      for (int q8 = 0; q8 < 8; ++q8)
        Bt[(c2 + q8) * 32 + r2] = hv[q8];
      async_join();
    }
    __syncthreads();

    v16h a0 = load_a_frag(&As[(wy * 32) * 32], 32);
    v16h a1 = load_a_frag(&As[(wy * 32 + 16) * 32], 32);
    v16h b0 = load_b_frag_c(&Bt[(wx * 32) * 32], 32);
    v16h b1 = load_b_frag_c(&Bt[(wx * 32 + 16) * 32], 32);
    acc[0][0] = wmma_f16(a0, b0, acc[0][0]);
    acc[0][1] = wmma_f16(a0, b1, acc[0][1]);
    acc[1][0] = wmma_f16(a1, b0, acc[1][0]);
    acc[1][1] = wmma_f16(a1, b1, acc[1][1]);
  }

  const int halfadd = (lane >> 4) << 3;
  const int ncol    = lane & 15;
#pragma unroll
  for (int ft = 0; ft < 2; ++ft)
#pragma unroll
    for (int fn = 0; fn < 2; ++fn) {
      const int col = col0 + wx * 32 + fn * 16 + ncol;
      const float bias = bo[col];
#pragma unroll
      for (int e = 0; e < 8; ++e) {
        const int m = row0 + wy * 32 + ft * 16 + e + halfadd;
        out[(size_t)m * 512 + col] = acc[ft][fn][e] + bias;
      }
    }
}

// ---------------------------------------------------------------------------
extern "C" void kernel_launch(void* const* d_in, const int* in_sizes, int n_in,
                              void* d_out, int out_size, void* d_ws, size_t ws_size,
                              hipStream_t stream) {
  const float* x    = (const float*)d_in[0];
  const float* Wqkv = (const float*)d_in[1];
  const float* bqkv = (const float*)d_in[2];
  const float* Wo   = (const float*)d_in[3];
  const float* bo   = (const float*)d_in[4];
  const float* relb = (const float*)d_in[5];
  float* out = (float*)d_out;

  const size_t NELEM = (size_t)4 * 8192 * 512;  // 16,777,216
  _Float16* q   = (_Float16*)d_ws;
  _Float16* k   = q + NELEM;
  _Float16* vt  = k + NELEM;
  _Float16* ctx = vt + NELEM;

  dim3 blk(256);
  qkv_kernel<<<dim3(32768 / 128, 1536 / 64), blk, 0, stream>>>(x, Wqkv, bqkv, q, k, vt);
  attn_kernel<<<dim3(4 * 32 * 8), blk, 0, stream>>>(q, k, vt, relb, ctx);
  proj_kernel<<<dim3(32768 / 128, 512 / 64), blk, 0, stream>>>(ctx, Wo, bo, out);
}
